// GATv2CoolingClassifier_70720931496098
// MI455X (gfx1250) — matrix operand order repaired and checked by reading, not compiled, not run on hardware
//
#include <hip/hip_runtime.h>
#include <hip/hip_fp16.h>
#include <hip/hip_bf16.h>

typedef __attribute__((ext_vector_type(16))) _Float16 v16h;
typedef __attribute__((ext_vector_type(8)))  float    v8f;

#define N_HEADS 4
#define HID_C 64
#define HC 256
#define N_GRAPHS 64
#define LN_EPS 1e-5f

// ---------- monotone float<->uint encoding for atomic segment-max ----------
static __device__ __forceinline__ unsigned encF(float f) {
  int i = __float_as_int(f);
  return (i < 0) ? ~(unsigned)i : ((unsigned)i | 0x80000000u);
}
static __device__ __forceinline__ float decF(unsigned u) {
  int i = (u & 0x80000000u) ? (int)(u & 0x7fffffffu) : (int)(~u);
  return __int_as_float(i);
}

// ---------- weight pre-swizzle: f32 [K,256] -> f16 fragment order ----------
// dst flat = ((kt*16 + nt)*32 + lane)*16 + j ; value = W[kt*32 + (j>>3)*16 + (lane>>4)*8 + (j&7)][nt*16 + (lane&15)]
__global__ void convert_w_kernel(const float* __restrict__ W, _Float16* __restrict__ Wsw, int K) {
  int tid = blockIdx.x * blockDim.x + threadIdx.x;
  if (tid >= K * HC) return;
  int j    = tid & 15;
  int lane = (tid >> 4) & 31;
  int nt   = (tid >> 9) & 15;
  int kt   = tid >> 13;
  int k = kt * 32 + (j >> 3) * 16 + (lane >> 4) * 8 + (j & 7);
  int n = nt * 16 + (lane & 15);
  Wsw[tid] = (_Float16)W[k * HC + n];
}

// ---------- encoder: h16 = relu(x @ enc_w + enc_b), padded rows zeroed ----------
__global__ void encoder_kernel(const float* __restrict__ x, const float* __restrict__ w,
                               const float* __restrict__ b, _Float16* __restrict__ h16,
                               int N, int Npad) {
  int tid = blockIdx.x * blockDim.x + threadIdx.x;
  if (tid >= Npad * HID_C) return;
  int n = tid >> 6, c = tid & 63;
  float acc = 0.f;
  if (n < N) {
    acc = b[c];
#pragma unroll
    for (int k = 0; k < 8; ++k) acc += x[n * 8 + k] * w[k * HID_C + c];
    acc = fmaxf(acc, 0.f);
  }
  h16[tid] = (_Float16)acc;
}

// ---------- WMMA GEMM: C[M,256] = A16[Mpad,K] @ Bsw16[K,256] + bias ----------
union AFrag { v16h v; float4 q[2]; };

__global__ __launch_bounds__(256) void gemm_wmma_kernel(
    const _Float16* __restrict__ A, const _Float16* __restrict__ Bsw,
    const float* __restrict__ bias, float* __restrict__ C, int M, int K) {
  const int wave = threadIdx.x >> 5;     // 0..7
  const int lane = threadIdx.x & 31;
  const int mBase = blockIdx.x * 32 + (wave & 1) * 16;
  const int nBase = (wave >> 1) * 64;
  const int aRowIdx = mBase + (lane & 15);
  const int hi = lane >> 4;

  v8f acc[4] = {};

  const int ktCount = K >> 5;
  for (int kt = 0; kt < ktCount; ++kt) {
    AFrag af;
    const _Float16* aRow = A + (size_t)aRowIdx * K + kt * 32 + hi * 8;
    af.q[0] = *reinterpret_cast<const float4*>(aRow);        // K = kt*32 + hi*8 .. +7
    af.q[1] = *reinterpret_cast<const float4*>(aRow + 16);   // K = kt*32+16+hi*8 .. +7
#pragma unroll
    for (int i = 0; i < 4; ++i) {
      const int nt = (nBase >> 4) + i;
      const v16h bf = *reinterpret_cast<const v16h*>(
          Bsw + ((((size_t)kt * 16 + nt) * 32 + lane) << 4));
      acc[i] = __builtin_amdgcn_wmma_f32_16x16x32_f16(
          false, af.v, false, bf, (short)0, acc[i], false, false);
    }
  }

  // C/D layout: lanes 0-15 -> rows mBase+0..7 ; lanes 16-31 -> rows mBase+8..15
#pragma unroll
  for (int i = 0; i < 4; ++i) {
    const int col = nBase + i * 16 + (lane & 15);
    const float bv = bias[col];
#pragma unroll
    for (int r = 0; r < 8; ++r) {
      const int row = mBase + hi * 8 + r;
      if (row < M) C[(size_t)row * HC + col] = acc[i][r] + bv;
    }
  }
}

// ---------- per-layer init: out=bias broadcast; mx=0(enc -inf-ish); den=0 ----------
__global__ void init_layer_kernel(float* __restrict__ out, const float* __restrict__ bias,
                                  unsigned* __restrict__ mx, float* __restrict__ den,
                                  int Npad, int N) {
  int tid = blockIdx.x * blockDim.x + threadIdx.x;
  if (tid < Npad * HC) out[tid] = bias[tid & 255];
  if (tid < N * N_HEADS) { mx[tid] = 0u; den[tid] = 0.f; }
}

// ---------- edge logits: one wave per edge ----------
__global__ __launch_bounds__(256) void edge_logits_kernel(
    const float* __restrict__ xl, const float* __restrict__ xr,
    const float* __restrict__ ea, const float* __restrict__ we,
    const float* __restrict__ att, const int* __restrict__ src,
    const int* __restrict__ dst, float* __restrict__ logits,
    unsigned* __restrict__ mx, int E) {
  int e = blockIdx.x * 8 + (threadIdx.x >> 5);
  int lane = threadIdx.x & 31;
  if (e >= E) return;
  int s = src[e], d = dst[e];
  float eav = ea[e];
  const float* xls = xl + (size_t)s * HC;
  const float* xrd = xr + (size_t)d * HC;
#pragma unroll
  for (int hd = 0; hd < N_HEADS; ++hd) {
    float partial = 0.f;
#pragma unroll
    for (int q = 0; q < 2; ++q) {
      int c = hd * HID_C + lane + q * 32;
      float m = xls[c] + xrd[c] + eav * we[c];
      m = (m > 0.f) ? m : 0.2f * m;
      partial += att[c] * m;
    }
#pragma unroll
    for (int off = 16; off; off >>= 1) partial += __shfl_xor(partial, off);
    if (lane == 0) {
      logits[e * N_HEADS + hd] = partial;
      atomicMax(&mx[d * N_HEADS + hd], encF(partial));
    }
  }
}

// ---------- exp + denominator ----------
__global__ void edge_exp_kernel(float* __restrict__ logits, const int* __restrict__ dst,
                                const unsigned* __restrict__ mx, float* __restrict__ den, int E) {
  int tid = blockIdx.x * blockDim.x + threadIdx.x;
  if (tid >= E * N_HEADS) return;
  int e = tid >> 2, hd = tid & 3;
  int d = dst[e];
  float ex = expf(logits[tid] - decF(mx[d * N_HEADS + hd]));
  logits[tid] = ex;
  atomicAdd(&den[d * N_HEADS + hd], ex);
}

// ---------- weighted scatter: out[dst] += xl[src] * alpha ----------
__global__ __launch_bounds__(256) void edge_aggregate_kernel(
    const float* __restrict__ xl, const float* __restrict__ ex,
    const float* __restrict__ den, const int* __restrict__ src,
    const int* __restrict__ dst, float* __restrict__ out, int E) {
  int e = blockIdx.x * 8 + (threadIdx.x >> 5);
  int lane = threadIdx.x & 31;
  if (e >= E) return;
  int s = src[e], d = dst[e];
  float a[N_HEADS];
#pragma unroll
  for (int hd = 0; hd < N_HEADS; ++hd)
    a[hd] = ex[e * N_HEADS + hd] / den[d * N_HEADS + hd];
  const float* xls = xl + (size_t)s * HC;
  float* outd = out + (size_t)d * HC;
#pragma unroll
  for (int q = 0; q < 8; ++q) {
    int c = lane + q * 32;
    atomicAdd(&outd[c], xls[c] * a[c >> 6]);
  }
}

// ---------- finalize: relu in-place (f32) + write f16 copy, pad rows zeroed ----------
__global__ void finalize_kernel(float* __restrict__ out, _Float16* __restrict__ h16,
                                int N, int Npad) {
  int tid = blockIdx.x * blockDim.x + threadIdx.x;
  if (tid >= Npad * HC) return;
  int n = tid >> 8;
  float v = (n < N) ? fmaxf(out[tid], 0.f) : 0.f;
  out[tid] = v;
  h16[tid] = (_Float16)v;
}

// ---------- pooling ----------
__global__ void init_pool_kernel(float* __restrict__ g, float* __restrict__ counts) {
  int tid = blockIdx.x * blockDim.x + threadIdx.x;
  if (tid < N_GRAPHS * HC) g[tid] = 0.f;
  if (tid < N_GRAPHS) counts[tid] = 0.f;
}

__global__ void pool_kernel(const float* __restrict__ h, const int* __restrict__ batch,
                            float* __restrict__ g, float* __restrict__ counts, int N) {
  int tid = blockIdx.x * blockDim.x + threadIdx.x;
  if (tid >= N * HC) return;
  int n = tid >> 8, c = tid & 255;
  int b = batch[n];
  atomicAdd(&g[b * HC + c], h[tid]);
  if (c == 0) atomicAdd(&counts[b], 1.f);
}

// ---------- post MLP + LN + head: one block per graph ----------
__global__ __launch_bounds__(128) void mlp_head_kernel(
    const float* __restrict__ g, const float* __restrict__ counts,
    const float* __restrict__ p1w, const float* __restrict__ p1b,
    const float* __restrict__ lng, const float* __restrict__ lnb,
    const float* __restrict__ p2w, const float* __restrict__ p2b,
    const float* __restrict__ hw, const float* __restrict__ hb,
    float* __restrict__ out) {
  __shared__ float sg[HC];
  __shared__ float sz[128];
  __shared__ float s2[64];
  __shared__ float sred[8];
  int gi = blockIdx.x, t = threadIdx.x;
  float cnt = fmaxf(counts[gi], 1.f);
  sg[t]       = g[gi * HC + t] / cnt;
  sg[t + 128] = g[gi * HC + t + 128] / cnt;
  __syncthreads();
  float z = p1b[t];
  for (int c = 0; c < HC; ++c) z += sg[c] * p1w[c * 128 + t];
  float s = z, sq = z * z;
#pragma unroll
  for (int off = 16; off; off >>= 1) { s += __shfl_xor(s, off); sq += __shfl_xor(sq, off); }
  if ((t & 31) == 0) { sred[t >> 5] = s; sred[4 + (t >> 5)] = sq; }
  __syncthreads();
  float mu  = (sred[0] + sred[1] + sred[2] + sred[3]) * (1.f / 128.f);
  float ms  = (sred[4] + sred[5] + sred[6] + sred[7]) * (1.f / 128.f);
  float var = ms - mu * mu;
  z = (z - mu) * rsqrtf(var + LN_EPS) * lng[t] + lnb[t];
  sz[t] = fmaxf(z, 0.f);
  __syncthreads();
  if (t < 64) {
    float z2 = p2b[t];
    for (int j = 0; j < 128; ++j) z2 += sz[j] * p2w[j * 64 + t];
    s2[t] = fmaxf(z2, 0.f);
  }
  __syncthreads();
  if (t == 0) {
    float acc = hb[0];
    for (int j = 0; j < 64; ++j) acc += s2[j] * hw[j];
    out[gi] = acc;
  }
}

// =====================================================================
extern "C" void kernel_launch(void* const* d_in, const int* in_sizes, int n_in,
                              void* d_out, int out_size, void* d_ws, size_t ws_size,
                              hipStream_t stream) {
  const int N = in_sizes[0] / 8;
  const int E = in_sizes[1];
  const int Npad = (N + 31) & ~31;

  const float* x        = (const float*)d_in[0];
  const float* ea       = (const float*)d_in[1];
  const int*   ei       = (const int*)d_in[2];
  const int*   batch    = (const int*)d_in[3];
  const float* enc_w    = (const float*)d_in[4];
  const float* enc_b    = (const float*)d_in[5];
  const float* g1_wl    = (const float*)d_in[6];
  const float* g1_bl    = (const float*)d_in[7];
  const float* g1_wr    = (const float*)d_in[8];
  const float* g1_br    = (const float*)d_in[9];
  const float* g1_we    = (const float*)d_in[10];
  const float* g1_att   = (const float*)d_in[11];
  const float* g1_bias  = (const float*)d_in[12];
  const float* g2_wl    = (const float*)d_in[13];
  const float* g2_bl    = (const float*)d_in[14];
  const float* g2_wr    = (const float*)d_in[15];
  const float* g2_br    = (const float*)d_in[16];
  const float* g2_we    = (const float*)d_in[17];
  const float* g2_att   = (const float*)d_in[18];
  const float* g2_bias  = (const float*)d_in[19];
  const float* p1_w     = (const float*)d_in[20];
  const float* p1_b     = (const float*)d_in[21];
  const float* ln_g     = (const float*)d_in[22];
  const float* ln_b     = (const float*)d_in[23];
  const float* p2_w     = (const float*)d_in[24];
  const float* p2_b     = (const float*)d_in[25];
  const float* head_w   = (const float*)d_in[26];
  const float* head_b   = (const float*)d_in[27];
  const int* src = ei;
  const int* dst = ei + E;
  float* out = (float*)d_out;

  // workspace carve-up (256B aligned)
  char* ws = (char*)d_ws;
  size_t o = 0;
  auto carve = [&](size_t bytes) { size_t r = o; o += (bytes + 255) & ~(size_t)255; return r; };
  _Float16* h16  = (_Float16*)(ws + carve((size_t)Npad * HC * 2));
  _Float16* w1l  = (_Float16*)(ws + carve((size_t)64 * HC * 2));
  _Float16* w1r  = (_Float16*)(ws + carve((size_t)64 * HC * 2));
  _Float16* w2l  = (_Float16*)(ws + carve((size_t)HC * HC * 2));
  _Float16* w2r  = (_Float16*)(ws + carve((size_t)HC * HC * 2));
  float*    xl   = (float*)(ws + carve((size_t)Npad * HC * 4));
  float*    xr   = (float*)(ws + carve((size_t)Npad * HC * 4));
  float*    lg   = (float*)(ws + carve((size_t)E * N_HEADS * 4));
  unsigned* mx   = (unsigned*)(ws + carve((size_t)N * N_HEADS * 4));
  float*    den  = (float*)(ws + carve((size_t)N * N_HEADS * 4));
  float*    agg  = (float*)(ws + carve((size_t)Npad * HC * 4));
  float*    gbuf = (float*)(ws + carve((size_t)N_GRAPHS * HC * 4));
  float*    cnts = (float*)(ws + carve((size_t)N_GRAPHS * 4));
  (void)ws_size;

  const int TB = 256;
  auto blk = [](long long n, int t) { return (unsigned)((n + t - 1) / t); };

  // 0) pre-swizzle weights -> f16 fragment layout
  convert_w_kernel<<<blk(64 * HC, TB), TB, 0, stream>>>(g1_wl, w1l, 64);
  convert_w_kernel<<<blk(64 * HC, TB), TB, 0, stream>>>(g1_wr, w1r, 64);
  convert_w_kernel<<<blk(HC * HC, TB), TB, 0, stream>>>(g2_wl, w2l, HC);
  convert_w_kernel<<<blk(HC * HC, TB), TB, 0, stream>>>(g2_wr, w2r, HC);

  // 1) encoder
  encoder_kernel<<<blk((long long)Npad * HID_C, TB), TB, 0, stream>>>(x, enc_w, enc_b, h16, N, Npad);

  // ===== GAT layer 1 (K=64) =====
  gemm_wmma_kernel<<<Npad / 32, 256, 0, stream>>>(h16, w1l, g1_bl, xl, N, 64);
  gemm_wmma_kernel<<<Npad / 32, 256, 0, stream>>>(h16, w1r, g1_br, xr, N, 64);
  init_layer_kernel<<<blk((long long)Npad * HC, TB), TB, 0, stream>>>(agg, g1_bias, mx, den, Npad, N);
  edge_logits_kernel<<<blk(E, 8), 256, 0, stream>>>(xl, xr, ea, g1_we, g1_att, src, dst, lg, mx, E);
  edge_exp_kernel<<<blk((long long)E * N_HEADS, TB), TB, 0, stream>>>(lg, dst, mx, den, E);
  edge_aggregate_kernel<<<blk(E, 8), 256, 0, stream>>>(xl, lg, den, src, dst, agg, E);
  finalize_kernel<<<blk((long long)Npad * HC, TB), TB, 0, stream>>>(agg, h16, N, Npad);

  // ===== GAT layer 2 (K=256) =====
  gemm_wmma_kernel<<<Npad / 32, 256, 0, stream>>>(h16, w2l, g2_bl, xl, N, HC);
  gemm_wmma_kernel<<<Npad / 32, 256, 0, stream>>>(h16, w2r, g2_br, xr, N, HC);
  init_layer_kernel<<<blk((long long)Npad * HC, TB), TB, 0, stream>>>(agg, g2_bias, mx, den, Npad, N);
  edge_logits_kernel<<<blk(E, 8), 256, 0, stream>>>(xl, xr, ea, g2_we, g2_att, src, dst, lg, mx, E);
  edge_exp_kernel<<<blk((long long)E * N_HEADS, TB), TB, 0, stream>>>(lg, dst, mx, den, E);
  edge_aggregate_kernel<<<blk(E, 8), 256, 0, stream>>>(xl, lg, den, src, dst, agg, E);
  finalize_kernel<<<blk((long long)Npad * HC, TB), TB, 0, stream>>>(agg, h16, N, Npad);

  // ===== pool + MLP head =====
  init_pool_kernel<<<blk(N_GRAPHS * HC, TB), TB, 0, stream>>>(gbuf, cnts);
  pool_kernel<<<blk((long long)N * HC, TB), TB, 0, stream>>>(agg, batch, gbuf, cnts, N);
  mlp_head_kernel<<<N_GRAPHS, 128, 0, stream>>>(gbuf, cnts, p1_w, p1_b, ln_g, ln_b,
                                                p2_w, p2_b, head_w, head_b, out);
}
